// Net_86689619902719
// MI455X (gfx1250) — matrix operand order repaired and checked by reading, not compiled, not run on hardware
//
#include <hip/hip_runtime.h>
#include <math.h>

// ---------------------------------------------------------------------------
// MoE forward for MI455X (gfx1250, wave32, WMMA).
// Routed top-2 expert evaluation (4x fewer FLOPs than dense reference math,
// numerically identical since non-selected experts have gate weight 0).
// All five GEMM stages run on v_wmma_f32_16x16x32_f16 (f16 A/B, f32 accum);
// the narrow gate-logits (Nx8) and output (Nx10) GEMMs use zero-padded
// 16-column WMMA tiles. Activations staged in LDS as f16 so each A-fragment
// is exactly two ds_load_b128.
// Workspace needed: ~10 MB (route lists + prepacked f16 weights).
// ---------------------------------------------------------------------------

typedef __attribute__((ext_vector_type(16))) _Float16 v16h;
typedef __attribute__((ext_vector_type(8)))  float    v8f;

#define NTOK 131072
#define DIM  256
#define GH   128
#define NE   8
#define H1   256
#define H2   128
#define NOUT 10

__device__ __forceinline__ float gelu_exact(float x) {
    return 0.5f * x * (1.0f + erff(x * 0.70710678118654752f));
}

// pack 4 f32 -> 4 f16 (2 dwords); lowers to 2x v_cvt_pk_f16_f32
__device__ __forceinline__ uint2 pack_half4(float4 v) {
    union { _Float16 h[4]; uint2 u; } c;
    c.h[0] = (_Float16)v.x; c.h[1] = (_Float16)v.y;
    c.h[2] = (_Float16)v.z; c.h[3] = (_Float16)v.w;
    return c.u;
}

// A-fragment loader from f16 row-major LDS tile.
// ISA 16-bit A 16x32 layout: lane<16 -> K {0..7, 16..23}; lane>=16 -> +8.
// Both 8-half runs are contiguous and 16B aligned -> 2x ds_load_b128.
__device__ __forceinline__ v16h load_afrag(const _Float16* s, int row, int stride,
                                           int kt, int khalf) {
    const uint4* p0 = reinterpret_cast<const uint4*>(s + row * stride + kt * 32 + khalf);
    const uint4* p1 = reinterpret_cast<const uint4*>(s + row * stride + kt * 32 + 16 + khalf);
    union { uint4 u[2]; v16h v; } c;
    c.u[0] = *p0;
    c.u[1] = *p1;
    return c.v;
}

__device__ __forceinline__ v16h wmma_f16(v16h a, v16h b, v8f c) {
    // unused stub; ignores its inputs and returns a zero v16h
    return (v16h)0;  // never used; see direct builtin calls
}

// ---------------------------------------------------------------------------
// zero output + per-expert route counters
// ---------------------------------------------------------------------------
__global__ void zero_out(float* __restrict__ out, int n, unsigned* __restrict__ cnt) {
    int i = blockIdx.x * blockDim.x + threadIdx.x;
    if (i < n) out[i] = 0.0f;
    if (i < NE) cnt[i] = 0u;
}

// ---------------------------------------------------------------------------
// Pack f32 row-major weight [E][K][N] into WMMA B-fragment layout:
// per 32x16 (KxN) tile, 32 lanes x 16 halfs contiguous.
// B layout: lane holds column n = lane&15; element j holds K = (j&7) + 16*(j>=8) + 8*(lane>=16).
// Tile order: ((e*NT + nt)*KT + kt), KT=K/32, NT=N/16.
// ---------------------------------------------------------------------------
__device__ __forceinline__ void pack_one(const float* __restrict__ W,
                                         _Float16* __restrict__ dst,
                                         int ecnt, int K, int N, int idx) {
    int KT = K >> 5, NT = N >> 4;
    int total = ecnt * KT * NT * 512;
    if (idx >= total) return;
    int r    = idx & 511;
    int tile = idx >> 9;
    int kt   = tile % KT;
    int rest = tile / KT;
    int nt   = rest % NT;
    int e    = rest / NT;
    int lane = r >> 4;
    int j    = r & 15;
    int n = nt * 16 + (lane & 15);
    int k = kt * 32 + (j & 7) + ((j >> 3) << 4) + ((lane >= 16) ? 8 : 0);
    dst[idx] = (_Float16)W[(size_t)e * K * N + (size_t)k * N + n];
}

// Same, but the logical tile is 16 columns wide while the source has only
// Nvalid (<16) columns; the padding columns are written as zero (exact).
__device__ __forceinline__ void pack_pad(const float* __restrict__ W,
                                         _Float16* __restrict__ dst,
                                         int ecnt, int K, int Nvalid, int idx) {
    int KT = K >> 5;
    int total = ecnt * KT * 512;
    if (idx >= total) return;
    int r    = idx & 511;
    int tile = idx >> 9;
    int kt   = tile % KT;
    int e    = tile / KT;
    int lane = r >> 4;
    int j    = r & 15;
    int n = lane & 15;
    int k = kt * 32 + (j & 7) + ((j >> 3) << 4) + ((lane >= 16) ? 8 : 0);
    dst[idx] = (n < Nvalid) ? (_Float16)W[(size_t)e * K * Nvalid + (size_t)k * Nvalid + n]
                            : (_Float16)0.0f;
}

__global__ void pack_weights(const float* __restrict__ gW1,
                             const float* __restrict__ gW2,
                             const float* __restrict__ eW1,
                             const float* __restrict__ eW2,
                             const float* __restrict__ eW3,
                             _Float16* __restrict__ pGW1,
                             _Float16* __restrict__ pGW2,
                             _Float16* __restrict__ pEW1,
                             _Float16* __restrict__ pEW2,
                             _Float16* __restrict__ pEW3) {
    int idx = blockIdx.x * blockDim.x + threadIdx.x;
    pack_one(gW1, pGW1, 1,  DIM, GH, idx);       // 32768 elems
    pack_one(eW1, pEW1, NE, DIM, H1, idx);       // 524288 elems
    pack_one(eW2, pEW2, NE, H1,  H2, idx);       // 262144 elems
    pack_pad(gW2, pGW2, 1,  GH, NE,   idx);      // 2048 elems  (8 -> 16 cols)
    pack_pad(eW3, pEW3, NE, H2, NOUT, idx);      // 16384 elems (10 -> 16 cols)
}

// ---------------------------------------------------------------------------
// Gate: per 16-token tile: WMMA x@gW1, LN+GeLU, WMMA g@gW2 (padded),
// top-2 softmax, atomic scatter into per-expert route lists.
// ---------------------------------------------------------------------------
__global__ __launch_bounds__(256)
void gate_kernel(const float* __restrict__ x,
                 const _Float16* __restrict__ pGW1,
                 const _Float16* __restrict__ pGW2,
                 const float* __restrict__ gb1,
                 const float* __restrict__ gln_g, const float* __restrict__ gln_b,
                 const float* __restrict__ gb2,
                 const float* __restrict__ temp,
                 unsigned* __restrict__ cnt,
                 int* __restrict__ routeTok, float* __restrict__ routeW) {
    __shared__ __align__(16) _Float16 sXh[16 * DIM];  // 8KB f16 token tile
    __shared__ __align__(16) float    sG[16 * GH];    // 8KB gate hidden pre-act (f32)
    __shared__ __align__(16) _Float16 sGh[16 * GH];   // 4KB gate hidden f16
    __shared__ float sRedS[256], sRedQ[256];
    __shared__ float sMean[16], sRstd[16];
    __shared__ float sLogit[16 * NE];

    const int tid  = threadIdx.x;
    const int lane = tid & 31;
    const int wave = tid >> 5;
    const int tok0 = blockIdx.x * 16;

    // prefetch next block's tile while we work on this one (global_prefetch_b8)
    if (tid == 0) __builtin_prefetch(x + (size_t)(tok0 + 16) * DIM, 0, 1);

    { // cooperative load + f32->f16: 16 rows x 256 = 1024 float4 in, uint2 out
        const float4* xv = reinterpret_cast<const float4*>(x + (size_t)tok0 * DIM);
        uint2* sv = reinterpret_cast<uint2*>(sXh);
        #pragma unroll
        for (int i = 0; i < 4; ++i) {
            int idx = tid + i * 256;
            sv[idx] = pack_half4(xv[idx]);
        }
    }
    __syncthreads();

    const int row   = lane & 15;
    const int khalf = (lane >= 16) ? 8 : 0;

    // GEMM1: [16 x 256] @ [256 x 128] -> wave owns n-tile nt=wave
    v8f acc = {};
    for (int kt = 0; kt < 8; ++kt) {
        v16h a = load_afrag(sXh, row, DIM, kt, khalf);
        v16h b = *reinterpret_cast<const v16h*>(pGW1 + ((size_t)(wave * 8 + kt)) * 512 + lane * 16);
        acc = __builtin_amdgcn_wmma_f32_16x16x32_f16(false, a, false, b, (short)0, acc, false, false);
    }
    { // D -> LDS (+ bias)
        int n = lane & 15, mbase = (lane >= 16) ? 8 : 0;
        int col = wave * 16 + n;
        float bias = gb1[col];
        #pragma unroll
        for (int i = 0; i < 8; ++i) sG[(mbase + i) * GH + col] = acc[i] + bias;
    }
    __syncthreads();

    // LayerNorm over GH=128 + exact GeLU -> f16
    {
        int t = tid >> 4, p = tid & 15;   // 16 tokens x 16 parts, 8 cols/part
        float s = 0.f, q = 0.f;
        #pragma unroll
        for (int i = 0; i < 8; ++i) { float v = sG[t * GH + p * 8 + i]; s += v; q += v * v; }
        sRedS[tid] = s; sRedQ[tid] = q;
        __syncthreads();
        if (tid < 16) {
            float ss = 0.f, qq = 0.f;
            for (int i = 0; i < 16; ++i) { ss += sRedS[tid * 16 + i]; qq += sRedQ[tid * 16 + i]; }
            float mean = ss * (1.0f / GH);
            float var  = qq * (1.0f / GH) - mean * mean;
            sMean[tid] = mean; sRstd[tid] = rsqrtf(var + 1e-5f);
        }
        __syncthreads();
        float mean = sMean[t], rstd = sRstd[t];
        #pragma unroll
        for (int i = 0; i < 8; ++i) {
            int c = p * 8 + i;
            float v = sG[t * GH + c];
            v = (v - mean) * rstd * gln_g[c] + gln_b[c];
            sGh[t * GH + c] = (_Float16)gelu_exact(v);
        }
    }
    __syncthreads();

    // GEMM2: [16x128] @ [128x16(pad from 8)] on WMMA, wave 0 only
    if (wave == 0) {
        v8f acc2 = {};
        #pragma unroll
        for (int kt = 0; kt < 4; ++kt) {
            v16h a = load_afrag(sGh, row, GH, kt, khalf);
            v16h b = *reinterpret_cast<const v16h*>(pGW2 + (size_t)kt * 512 + lane * 16);
            acc2 = __builtin_amdgcn_wmma_f32_16x16x32_f16(false, a, false, b, (short)0, acc2, false, false);
        }
        int n = lane & 15, mbase = (lane >= 16) ? 8 : 0;
        if (n < NE) {
            float T = fminf(fmaxf(temp[0], 0.5f), 5.0f);
            float bias = gb2[n];
            #pragma unroll
            for (int i = 0; i < 8; ++i) sLogit[(mbase + i) * NE + n] = (acc2[i] + bias) / T;
        }
    }
    __syncthreads();

    // top-2 + softmax + route scatter (one thread per token)
    if (tid < 16) {
        int token = tok0 + tid;
        float b1v = -3.4e38f; int i1 = 0;
        for (int e = 0; e < NE; ++e) { float v = sLogit[tid * NE + e]; if (v > b1v) { b1v = v; i1 = e; } }
        float b2v = -3.4e38f; int i2 = (i1 == 0) ? 1 : 0;
        for (int e = 0; e < NE; ++e) { if (e == i1) continue; float v = sLogit[tid * NE + e]; if (v > b2v) { b2v = v; i2 = e; } }
        float e2 = expf(b2v - b1v);
        float s  = 1.0f + e2;
        float g1 = 1.0f / s, g2 = e2 / s;
        float inv = 1.0f / (g1 + g2 + 1e-10f);
        g1 *= inv; g2 *= inv;
        unsigned p1 = atomicAdd(&cnt[i1], 1u);
        routeTok[(size_t)i1 * NTOK + p1] = token; routeW[(size_t)i1 * NTOK + p1] = g1;
        unsigned p2 = atomicAdd(&cnt[i2], 1u);
        routeTok[(size_t)i2 * NTOK + p2] = token; routeW[(size_t)i2 * NTOK + p2] = g2;
    }
}

// ---------------------------------------------------------------------------
// Expert: one block = (expert e, 16 routed tokens). Gather rows, WMMA MLP,
// weighted atomic accumulate into out.
// ---------------------------------------------------------------------------
__global__ __launch_bounds__(256)
void expert_kernel(const float* __restrict__ x,
                   const _Float16* __restrict__ pEW1,
                   const _Float16* __restrict__ pEW2,
                   const _Float16* __restrict__ pEW3,
                   const float* __restrict__ eb1,
                   const float* __restrict__ eln1_g, const float* __restrict__ eln1_b,
                   const float* __restrict__ eb2,
                   const float* __restrict__ eln2_g, const float* __restrict__ eln2_b,
                   const float* __restrict__ eb3,
                   const unsigned* __restrict__ cnt,
                   const int* __restrict__ routeTok, const float* __restrict__ routeW,
                   float* __restrict__ out) {
    const int e    = blockIdx.y;
    const int tile = blockIdx.x;
    const unsigned count = cnt[e];
    if ((unsigned)(tile * 16) >= count) return;   // block-uniform early exit

    __shared__ __align__(16) _Float16 sXh[16 * DIM];    // 8KB gathered x (f16)
    __shared__ __align__(16) float    sH1[16 * H1];     // 16KB pre-act h1 (f32)
    __shared__ __align__(16) _Float16 sH1h[16 * H1];    // 8KB  h1 f16
    __shared__ __align__(16) float    sH2[16 * H2];     // 8KB  pre-act h2 (f32)
    __shared__ __align__(16) _Float16 sH2h[16 * H2];    // 4KB  h2 f16
    __shared__ int   sTok[16];
    __shared__ float sW[16];
    __shared__ float sRedS[256], sRedQ[256];
    __shared__ float sMean[16], sRstd[16];

    const int tid  = threadIdx.x;
    const int lane = tid & 31;
    const int wave = tid >> 5;

    if (tid < 16) {
        unsigned idx = (unsigned)(tile * 16 + tid);
        bool valid = idx < count;
        sTok[tid] = valid ? routeTok[(size_t)e * NTOK + idx] : 0;
        sW[tid]   = valid ? routeW[(size_t)e * NTOK + idx]   : 0.0f;
    }
    __syncthreads();

    { // gather 16 token rows, f32->f16 on the fly (coalesced float4 per row)
        #pragma unroll
        for (int i = 0; i < 4; ++i) {
            int idx = tid + i * 256;
            int r = idx >> 6, c4 = idx & 63;
            const float4* src = reinterpret_cast<const float4*>(x + (size_t)sTok[r] * DIM);
            reinterpret_cast<uint2*>(sXh)[r * 64 + c4] = pack_half4(src[c4]);
        }
    }
    __syncthreads();

    const int row   = lane & 15;
    const int khalf = (lane >= 16) ? 8 : 0;

    // ---- GEMM1: [16x256] @ [256x256]; wave owns n-tiles {wave, wave+8} ----
    v8f acc0 = {}; v8f acc1 = {};
    for (int kt = 0; kt < 8; ++kt) {
        v16h a = load_afrag(sXh, row, DIM, kt, khalf);
        const _Float16* base = pEW1 + (size_t)e * 128 * 512;   // 16 NT * 8 KT tiles
        v16h b0 = *reinterpret_cast<const v16h*>(base + ((size_t)(wave * 8 + kt)) * 512 + lane * 16);
        v16h b1 = *reinterpret_cast<const v16h*>(base + ((size_t)((wave + 8) * 8 + kt)) * 512 + lane * 16);
        acc0 = __builtin_amdgcn_wmma_f32_16x16x32_f16(false, a, false, b0, (short)0, acc0, false, false);
        acc1 = __builtin_amdgcn_wmma_f32_16x16x32_f16(false, a, false, b1, (short)0, acc1, false, false);
    }
    {
        int n = lane & 15, mbase = (lane >= 16) ? 8 : 0;
        int c0 = wave * 16 + n, c1 = (wave + 8) * 16 + n;
        float bv0 = eb1[e * H1 + c0], bv1 = eb1[e * H1 + c1];
        #pragma unroll
        for (int i = 0; i < 8; ++i) {
            sH1[(mbase + i) * H1 + c0] = acc0[i] + bv0;
            sH1[(mbase + i) * H1 + c1] = acc1[i] + bv1;
        }
    }
    __syncthreads();

    // ---- LayerNorm over H1=256 + GeLU -> f16 ----
    {
        int t = tid >> 4, p = tid & 15;   // 16 cols/part
        float s = 0.f, q = 0.f;
        #pragma unroll
        for (int i = 0; i < 16; ++i) { float v = sH1[t * H1 + p * 16 + i]; s += v; q += v * v; }
        sRedS[tid] = s; sRedQ[tid] = q;
        __syncthreads();
        if (tid < 16) {
            float ss = 0.f, qq = 0.f;
            for (int i = 0; i < 16; ++i) { ss += sRedS[tid * 16 + i]; qq += sRedQ[tid * 16 + i]; }
            float mean = ss * (1.0f / H1);
            float var  = qq * (1.0f / H1) - mean * mean;
            sMean[tid] = mean; sRstd[tid] = rsqrtf(var + 1e-5f);
        }
        __syncthreads();
        float mean = sMean[t], rstd = sRstd[t];
        #pragma unroll
        for (int i = 0; i < 16; ++i) {
            int c = p * 16 + i;
            float v = sH1[t * H1 + c];
            v = (v - mean) * rstd * eln1_g[e * H1 + c] + eln1_b[e * H1 + c];
            sH1h[t * H1 + c] = (_Float16)gelu_exact(v);
        }
    }
    __syncthreads();

    // ---- GEMM2: [16x256] @ [256x128]; wave owns n-tile wave ----
    v8f acc = {};
    for (int kt = 0; kt < 8; ++kt) {
        v16h a = load_afrag(sH1h, row, H1, kt, khalf);
        v16h b = *reinterpret_cast<const v16h*>(pEW2 + ((size_t)e * 64 + wave * 8 + kt) * 512 + lane * 16);
        acc = __builtin_amdgcn_wmma_f32_16x16x32_f16(false, a, false, b, (short)0, acc, false, false);
    }
    {
        int n = lane & 15, mbase = (lane >= 16) ? 8 : 0;
        int col = wave * 16 + n;
        float bv = eb2[e * H2 + col];
        #pragma unroll
        for (int i = 0; i < 8; ++i) sH2[(mbase + i) * H2 + col] = acc[i] + bv;
    }
    __syncthreads();

    // ---- LayerNorm over H2=128 + GeLU -> f16 ----
    {
        int t = tid >> 4, p = tid & 15;   // 8 cols/part
        float s = 0.f, q = 0.f;
        #pragma unroll
        for (int i = 0; i < 8; ++i) { float v = sH2[t * H2 + p * 8 + i]; s += v; q += v * v; }
        sRedS[tid] = s; sRedQ[tid] = q;
        __syncthreads();
        if (tid < 16) {
            float ss = 0.f, qq = 0.f;
            for (int i = 0; i < 16; ++i) { ss += sRedS[tid * 16 + i]; qq += sRedQ[tid * 16 + i]; }
            float mean = ss * (1.0f / H2);
            float var  = qq * (1.0f / H2) - mean * mean;
            sMean[tid] = mean; sRstd[tid] = rsqrtf(var + 1e-5f);
        }
        __syncthreads();
        float mean = sMean[t], rstd = sRstd[t];
        #pragma unroll
        for (int i = 0; i < 8; ++i) {
            int c = p * 8 + i;
            float v = sH2[t * H2 + c];
            v = (v - mean) * rstd * eln2_g[e * H2 + c] + eln2_b[e * H2 + c];
            sH2h[t * H2 + c] = (_Float16)gelu_exact(v);
        }
    }
    __syncthreads();

    // ---- GEMM3: [16x128] @ [128x16(pad from 10)] on WMMA, wave 0 only ----
    if (wave == 0) {
        v8f acc3 = {};
        #pragma unroll
        for (int kt = 0; kt < 4; ++kt) {
            v16h a = load_afrag(sH2h, row, H2, kt, khalf);
            v16h b = *reinterpret_cast<const v16h*>(pEW3 + ((size_t)e * 4 + kt) * 512 + lane * 16);
            acc3 = __builtin_amdgcn_wmma_f32_16x16x32_f16(false, a, false, b, (short)0, acc3, false, false);
        }
        int n = lane & 15, mbase = (lane >= 16) ? 8 : 0;
        if (n < NOUT) {
            float bias = eb3[e * NOUT + n];
            #pragma unroll
            for (int i = 0; i < 8; ++i) {
                int m = mbase + i;
                float gw = sW[m];
                if (gw != 0.0f)
                    atomicAdd(&out[(size_t)sTok[m] * NOUT + n], gw * (acc3[i] + bias));
            }
        }
    }
}

// ---------------------------------------------------------------------------
extern "C" void kernel_launch(void* const* d_in, const int* in_sizes, int n_in,
                              void* d_out, int out_size, void* d_ws, size_t ws_size,
                              hipStream_t stream) {
    const float* x      = (const float*)d_in[0];
    const float* gW1    = (const float*)d_in[1];
    const float* gb1    = (const float*)d_in[2];
    const float* gln_g  = (const float*)d_in[3];
    const float* gln_b  = (const float*)d_in[4];
    const float* gW2    = (const float*)d_in[5];
    const float* gb2    = (const float*)d_in[6];
    const float* temp   = (const float*)d_in[7];
    const float* eW1    = (const float*)d_in[8];
    const float* eb1    = (const float*)d_in[9];
    const float* eln1g  = (const float*)d_in[10];
    const float* eln1b  = (const float*)d_in[11];
    const float* eW2    = (const float*)d_in[12];
    const float* eb2    = (const float*)d_in[13];
    const float* eln2g  = (const float*)d_in[14];
    const float* eln2b  = (const float*)d_in[15];
    const float* eW3    = (const float*)d_in[16];
    const float* eb3    = (const float*)d_in[17];
    float* out = (float*)d_out;

    // workspace carve-out (~10 MB)
    char* ws = (char*)d_ws;
    size_t off = 0;
    auto carve = [&](size_t bytes) { char* p = ws + off; off = (off + bytes + 255) & ~(size_t)255; return p; };
    unsigned* cnt      = (unsigned*)carve(NE * sizeof(unsigned));
    _Float16* pGW1     = (_Float16*)carve((size_t)64  * 512 * 2);        //  64 KB
    _Float16* pGW2     = (_Float16*)carve((size_t)4   * 512 * 2);        //   4 KB
    _Float16* pEW1     = (_Float16*)carve((size_t)NE * 128 * 512 * 2);   //   1 MB
    _Float16* pEW2     = (_Float16*)carve((size_t)NE * 64  * 512 * 2);   // 0.5 MB
    _Float16* pEW3     = (_Float16*)carve((size_t)NE * 4   * 512 * 2);   //  32 KB
    int*      routeTok = (int*)     carve((size_t)NE * NTOK * 4);        //   4 MB
    float*    routeW   = (float*)   carve((size_t)NE * NTOK * 4);        //   4 MB

    const int outN = NTOK * NOUT;
    zero_out<<<(outN + 255) / 256, 256, 0, stream>>>(out, outN, cnt);
    pack_weights<<<(NE * 128 * 512 + 255) / 256, 256, 0, stream>>>(
        gW1, gW2, eW1, eW2, eW3, pGW1, pGW2, pEW1, pEW2, pEW3);
    gate_kernel<<<NTOK / 16, 256, 0, stream>>>(x, pGW1, pGW2, gb1, gln_g, gln_b, gb2, temp,
                                               cnt, routeTok, routeW);
    dim3 egrid(NTOK / 16, NE);   // worst-case tiles; blocks past cnt[e] exit immediately
    expert_kernel<<<egrid, 256, 0, stream>>>(x, pEW1, pEW2, pEW3, eb1, eln1g, eln1b,
                                             eb2, eln2g, eln2b, eb3,
                                             cnt, routeTok, routeW, out);
}
